// ReinforceLayer_9431748182614
// MI455X (gfx1250) — matrix operand order repaired and checked by reading, not compiled, not run on hardware
//
#include <hip/hip_runtime.h>
#include <hip/hip_bf16.h>
#include <math.h>
#include <stdint.h>

// ---------------------------------------------------------------------------
// Problem constants (from the reference)
// ---------------------------------------------------------------------------
constexpr int NB   = 128;   // batch
constexpr int CIc  = 3;     // input channels
constexpr int HIc  = 128;
constexpr int WIc  = 128;
constexpr int Gc   = 12;    // glimpses
constexpr int HGc  = 16;
constexpr int WGc  = 16;
constexpr int HIDLIN = 512;
constexpr int HIDc = (HIc / 8) * (WIc / 8) * 128;   // 32768

// ---------------------------------------------------------------------------
// WMMA types (CDNA5 / gfx1250, wave32)
// ---------------------------------------------------------------------------
typedef __bf16 v16bf __attribute__((ext_vector_type(16)));
typedef __bf16 v8bf  __attribute__((ext_vector_type(8)));
typedef float  v8f   __attribute__((ext_vector_type(8)));
typedef unsigned v4u __attribute__((ext_vector_type(4)));
typedef unsigned v8u __attribute__((ext_vector_type(8)));

__device__ __forceinline__ __bf16 to_bf16(float f)   { return (__bf16)f; }
__device__ __forceinline__ float  bf16_to_f32(__bf16 h) { return (float)h; }
__device__ __forceinline__ __bf16 bf16_zero() {
    return __builtin_bit_cast(__bf16, (unsigned short)0);
}

// ---------------------------------------------------------------------------
// gfx1250 async global->LDS copy (ASYNCcnt path).  Low 32 bits of a generic
// pointer into LDS are the LDS byte offset (ISA 10.2 aperture truncation).
// ---------------------------------------------------------------------------
__device__ __forceinline__ void async_b64(void* lds_ptr, const void* gptr) {
    unsigned lds_off = (unsigned)(unsigned long long)(uintptr_t)lds_ptr;
    unsigned long long ga = (unsigned long long)(uintptr_t)gptr;
    asm volatile("global_load_async_to_lds_b64 %0, %1, off"
                 :: "v"(lds_off), "v"(ga) : "memory");
}
__device__ __forceinline__ void wait_async0() {
    asm volatile("s_wait_asynccnt 0x0" ::: "memory");
}

// ---------------------------------------------------------------------------
// gfx1250 Tensor Data Mover: one instruction DMAs a 2D tile (tileM rows x
// tileK bf16) from global (row stride rowStrideElems) into LDS at lds_off,
// padding each 64-byte row with 16 bytes so LDS row stride = 40 elems.
// D# layout per ISA 8.3/8.4 (group0: count/lds/global/type=2; group1:
// data_size=2B, pad_enable, pad_interval=16 DW, pad_amount=4 DW, dims).
// ---------------------------------------------------------------------------
__device__ __forceinline__ void tdm_load_2d_pad(unsigned lds_off, const void* gptr,
                                                unsigned tileK, unsigned tileM,
                                                unsigned long long rowStrideElems) {
    unsigned long long ga = (unsigned long long)(uintptr_t)gptr;
    v4u g0; v8u g1;
    g0[0] = 1u;                                        // count=1 (user D#)
    g0[1] = lds_off;                                   // lds_addr (bytes)
    g0[2] = (unsigned)(ga & 0xFFFFFFFFu);              // global_addr[31:0]
    g0[3] = (unsigned)((ga >> 32) & 0x01FFFFFFu)       // global_addr[56:32]
          | (2u << 30);                                // type = 2 ("image")
    const unsigned pad_interval = 3u;                  // 16 DWORDs (64 B)
    const unsigned pad_amount   = 3u;                  // 4 DWORDs  (16 B)
    g1[0] = (1u << 16)                                 // data_size = 2 bytes
          | (1u << 20)                                 // pad_enable
          | (pad_interval << 22) | (pad_amount << 25);
    g1[1] = (tileK & 0xFFFFu) << 16;                   // tensor_dim0[15:0]
    g1[2] = ((tileK >> 16) & 0xFFFFu)                  // tensor_dim0[31:16]
          | ((tileM & 0xFFFFu) << 16);                 // tensor_dim1[15:0]
    g1[3] = ((tileM >> 16) & 0xFFFFu)                  // tensor_dim1[31:16]
          | ((tileK & 0xFFFFu) << 16);                 // tile_dim0
    g1[4] = (tileM & 0xFFFFu);                         // tile_dim1 (tile_dim2=0)
    g1[5] = (unsigned)(rowStrideElems & 0xFFFFFFFFu);  // dim0_stride[31:0]
    g1[6] = (unsigned)((rowStrideElems >> 32) & 0xFFFFu); // dim0_stride[47:32]
    g1[7] = 0u;
    asm volatile("tensor_load_to_lds %0, %1" :: "s"(g0), "s"(g1) : "memory");
}

// ---------------------------------------------------------------------------
// Vectorized fragment loads: every fragment = two 16-byte LDS loads.
//  A tile: row-major [m][k], row stride LDT=40 elems (80 B, 16B aligned).
//    lane L: m=L%16, kb=(L<16?0:8): runs [kb,kb+8) and [kb+16,kb+24).
//  B tile: transposed [n][k], same stride.
//    lane L: n=L%16, kb=(L<16?0:8): runs [kb,kb+8) and [kb+16,kb+24).
// ---------------------------------------------------------------------------
constexpr int LDT = 40;

__device__ __forceinline__ v16bf cat8(v8bf lo, v8bf hi) {
    return __builtin_shufflevector(lo, hi, 0,1,2,3,4,5,6,7,8,9,10,11,12,13,14,15);
}
__device__ __forceinline__ v16bf load_a_frag(const __bf16* tile, int lane) {
    const int m  = lane & 15;
    const int kb = (lane < 16) ? 0 : 8;
    const __bf16* r = tile + m * LDT + kb;
    return cat8(*(const v8bf*)r, *(const v8bf*)(r + 16));
}
__device__ __forceinline__ v16bf load_b_frag_t(const __bf16* tileT, int lane) {
    const int n  = lane & 15;
    const int kb = (lane < 16) ? 0 : 8;
    const __bf16* r = tileT + n * LDT + kb;
    return cat8(*(const v8bf*)r, *(const v8bf*)(r + 16));
}

// ---------------------------------------------------------------------------
// Conv3x3 (pad 1, stride 1) + bias + ReLU, implicit GEMM, bf16 in/out.
// Block = 256 thr (8 waves): tile = 32 cout x 256 pixels.
// Wave = 16 cout x 64 pixels = 4 accumulators -> 4 WMMA per K-step.
// Grid: (H*W/256, COUT/32, NB)
// ---------------------------------------------------------------------------
template <int CIN, int COUT, int H, int W>
__global__ __launch_bounds__(256)
void conv3x3_relu_wmma(const __bf16* __restrict__ x,   // [NB][CIN][H][W]
                       const __bf16* __restrict__ w,   // [COUT][CIN*9]
                       const float*  __restrict__ bias,// [COUT] f32
                       __bf16* __restrict__ y)         // [NB][COUT][H][W]
{
    constexpr int KTOT = CIN * 9;
    constexpr int KPAD = (KTOT + 31) & ~31;
    constexpr int PIXT = 256;

    __shared__ __attribute__((aligned(16))) __bf16 sW [32   * LDT]; // [cout][k]
    __shared__ __attribute__((aligned(16))) __bf16 sXt[PIXT * LDT]; // [pix][k]

    const int tid  = threadIdx.x;
    const int lane = tid & 31;
    const int wv   = tid >> 5;       // 0..7
    const int wrow = wv >> 2;        // cout 16-subtile: 0..1
    const int wcol = wv & 3;         // pixel 64-subtile: 0..3

    const int pixBase  = blockIdx.x * PIXT;
    const int coutBase = blockIdx.y * 32;
    const int b        = blockIdx.z;

    // each thread owns one im2col column (its own pixel)
    const int pix = pixBase + tid;
    const int py  = pix / W;
    const int px  = pix - py * W;

    v8f acc[4];
#pragma unroll
    for (int j = 0; j < 4; ++j) acc[j] = v8f{0.f,0.f,0.f,0.f,0.f,0.f,0.f,0.f};

    for (int k0 = 0; k0 < KPAD; k0 += 32) {
        // ---- stage 32x32 weight tile (row-major [cout][k]) ----
        if (k0 + 32 <= KTOT) {
            // contiguous + 8B aligned: async DMA to LDS (ASYNCcnt path)
            const int co = tid >> 3;          // 0..31
            const int ch = tid & 7;           // 0..7  (4 bf16 per b64)
            async_b64(&sW[co * LDT + 4 * ch],
                      &w[(coutBase + co) * KTOT + k0 + 4 * ch]);
        } else {
            // ragged tail (conv1: KTOT=27): scalar with zero pad
#pragma unroll
            for (int i = tid; i < 32 * 32; i += 256) {
                const int co = i >> 5;
                const int kk = i & 31;
                const int k  = k0 + kk;
                sW[co * LDT + kk] =
                    (k < KTOT) ? w[(coutBase + co) * KTOT + k] : bf16_zero();
            }
        }
        // ---- stage transposed 256x32 im2col tile: one column per thread ----
        __bf16 col[32];
#pragma unroll
        for (int kk = 0; kk < 32; ++kk) {
            const int k = k0 + kk;
            __bf16 v = bf16_zero();
            if (k < KTOT) {
                const int cin = k / 9;
                const int r   = k - cin * 9;
                const int ky  = r / 3;
                const int kx  = r - ky * 3;
                const int iy  = py + ky - 1;
                const int ix  = px + kx - 1;
                if (iy >= 0 && iy < H && ix >= 0 && ix < W)
                    v = x[((b * CIN + cin) * H + iy) * W + ix];
            }
            col[kk] = v;
        }
#pragma unroll
        for (int q = 0; q < 4; ++q)   // 4 x ds_store_b128
            *(v8bf*)&sXt[tid * LDT + 8 * q] = *(const v8bf*)&col[8 * q];

        if (k0 + 32 <= KTOT) wait_async0();
        __syncthreads();

        if (k0 + 32 < KTOT)
            __builtin_prefetch(&w[(coutBase + (tid >> 3)) * KTOT + k0 + 32], 0, 1);

        const v16bf af = load_a_frag(&sW[wrow * 16 * LDT], lane);
#pragma unroll
        for (int j = 0; j < 4; ++j) {
            const v16bf bf = load_b_frag_t(&sXt[(wcol * 64 + j * 16) * LDT], lane);
            acc[j] = __builtin_amdgcn_wmma_f32_16x16x32_bf16(
                         false, af, false, bf, (short)0, acc[j], false, false);
        }
        __syncthreads();
    }

    // ---- epilogue: bias + relu, bf16 store, per C/D layout ----
    const int mbase = wrow * 16 + ((lane < 16) ? 0 : 8);
#pragma unroll
    for (int j = 0; j < 4; ++j) {
        const int n   = wcol * 64 + j * 16 + (lane & 15);
        const int opx = pixBase + n;
        const int oy  = opx / W;
        const int ox  = opx - oy * W;
#pragma unroll
        for (int v = 0; v < 8; ++v) {
            const int co = coutBase + mbase + v;
            float r = acc[j][v] + bias[co];
            r = r > 0.f ? r : 0.f;
            y[((b * COUT + co) * H + oy) * W + ox] = to_bf16(r);
        }
    }
}

// ---------------------------------------------------------------------------
// KxK max pooling (stride K), NCHW, bf16.
// ---------------------------------------------------------------------------
template <int C, int H, int W, int K>
__global__ void maxpool_kernel(const __bf16* __restrict__ x,
                               __bf16* __restrict__ y, int total)
{
    constexpr int HO = H / K, WO = W / K;
    const int idx = blockIdx.x * blockDim.x + threadIdx.x;
    if (idx >= total) return;
    int t = idx;
    const int wo = t % WO; t /= WO;
    const int ho = t % HO; t /= HO;
    const int c  = t % C;
    const int b  = t / C;
    const __bf16* base = &x[((b * C + c) * H + ho * K) * W + wo * K];
    float m = bf16_to_f32(base[0]);
#pragma unroll
    for (int i = 0; i < K; ++i)
#pragma unroll
        for (int j = 0; j < K; ++j)
            m = fmaxf(m, bf16_to_f32(base[i * W + j]));
    y[idx] = to_bf16(m);
}

// ---------------------------------------------------------------------------
// GEMM: Y[M][N] = relu?(A[M][K] * Bw[K][N] + bias[N]); A, Bw bf16.
// Block = 32 M x 128 N (8 waves); wave = 16 M x 32 N = 2 accumulators.
// A tile loaded with ONE tensor_load_to_lds (TDM, TENSORcnt path).
// B tile transposed in registers (per-k coalesced loads, b128 LDS stores).
// Grid: (ceil(N/128), M/32). K multiple of 32. N guarded (N=36 case).
// ---------------------------------------------------------------------------
template <bool RELU, bool OUTBF>
__global__ __launch_bounds__(256)
void gemm_bias_wmma(const __bf16* __restrict__ A,
                    const __bf16* __restrict__ Bw,
                    const float*  __restrict__ bias,
                    void* __restrict__ Yv,
                    int M, int K, int N)
{
    __shared__ __attribute__((aligned(16))) __bf16 sA [32  * LDT]; // [m][k]
    __shared__ __attribute__((aligned(16))) __bf16 sBt[128 * LDT]; // [n][k]

    const int tid  = threadIdx.x;
    const int lane = tid & 31;
    const int wv   = tid >> 5;
    const int wrow = wv >> 2;          // m 16-subtile
    const int wcol = wv & 3;           // n 32-subtile

    const int nBase = blockIdx.x * 128;
    const int mBase = blockIdx.y * 32;

    // B transpose-staging assignment: 2 threads per output column
    const int bn  = tid >> 1;          // 0..127
    const int bkb = (tid & 1) * 16;    // 0 or 16
    const unsigned sA_base = (unsigned)(uintptr_t)&sA[0];

    v8f acc[2] = { v8f{0.f,0.f,0.f,0.f,0.f,0.f,0.f,0.f},
                   v8f{0.f,0.f,0.f,0.f,0.f,0.f,0.f,0.f} };

    for (int k0 = 0; k0 < K; k0 += 32) {
        // ---- A tile: single TDM descriptor load (wave 0 issues) ----
        if (wv == 0) {
            tdm_load_2d_pad(sA_base, &A[(size_t)mBase * K + k0],
                            /*tileK=*/32, /*tileM=*/32,
                            /*rowStrideElems=*/(unsigned long long)K);
        }
        // ---- B tile: transpose in registers ----
        {
            __bf16 colv[16];
#pragma unroll
            for (int j = 0; j < 16; ++j) {
                const int nn = nBase + bn;
                colv[j] = (nn < N) ? Bw[(size_t)(k0 + bkb + j) * N + nn]
                                   : bf16_zero();
            }
            *(v8bf*)&sBt[bn * LDT + bkb]     = *(const v8bf*)&colv[0];
            *(v8bf*)&sBt[bn * LDT + bkb + 8] = *(const v8bf*)&colv[8];
        }
        if (wv == 0) __builtin_amdgcn_s_wait_tensorcnt(0);
        __syncthreads();

        if (k0 + 32 < K)
            __builtin_prefetch(&Bw[(size_t)(k0 + 32 + (tid >> 3)) * N + nBase], 0, 1);

        const v16bf af = load_a_frag(&sA[wrow * 16 * LDT], lane);
#pragma unroll
        for (int j = 0; j < 2; ++j) {
            const v16bf bf = load_b_frag_t(&sBt[(wcol * 32 + j * 16) * LDT], lane);
            acc[j] = __builtin_amdgcn_wmma_f32_16x16x32_bf16(
                         false, af, false, bf, (short)0, acc[j], false, false);
        }
        __syncthreads();
    }

    const int mb = mBase + wrow * 16 + ((lane < 16) ? 0 : 8);
#pragma unroll
    for (int j = 0; j < 2; ++j) {
        const int n = nBase + wcol * 32 + j * 16 + (lane & 15);
        if (n < N) {
            const float bn2 = bias[n];
#pragma unroll
            for (int v = 0; v < 8; ++v) {
                float r = acc[j][v] + bn2;
                if (RELU) r = r > 0.f ? r : 0.f;
                if (OUTBF) ((__bf16*)Yv)[(size_t)(mb + v) * N + n] = to_bf16(r);
                else       ((float*)Yv)[(size_t)(mb + v) * N + n] = r;
            }
        }
    }
}

// ---------------------------------------------------------------------------
// f32 -> bf16 conversion (image + weights, once per launch)
// ---------------------------------------------------------------------------
__global__ void f32_to_bf16_kernel(const float* __restrict__ src,
                                   __bf16* __restrict__ dst, int n)
{
    const int i = blockIdx.x * blockDim.x + threadIdx.x;
    if (i < n) dst[i] = to_bf16(src[i]);
}

// ---------------------------------------------------------------------------
// Sampling head: means / sigmas / sample / integer glimpse coordinates.
// ---------------------------------------------------------------------------
__global__ void sample_kernel(const float* __restrict__ prep,   // [NB][G*3] f32
                              const float* __restrict__ noise,  // [NB][G][2]
                              float* __restrict__ means,
                              float* __restrict__ sigmas,
                              float* __restrict__ sample,
                              int*   __restrict__ pts)
{
    const int i = blockIdx.x * blockDim.x + threadIdx.x;
    if (i >= NB * Gc) return;
    const int b = i / Gc;
    const int g = i - b * Gc;

    const float m0 = prep[b * (Gc * 3) + g * 3 + 0];
    const float m1 = prep[b * (Gc * 3) + g * 3 + 1];
    const float s  = prep[b * (Gc * 3) + g * 3 + 2] + 2.0f;   // SIGMA_BOOST
    const float sp   = fmaxf(s, 0.f) + log1pf(expf(-fabsf(s)));
    const float sig0 = sp * (float)HIc + 1e-7f;
    const float sig1 = sp * (float)WIc + 1e-7f;

    const float sa0 = m0 + sig0 * noise[i * 2 + 0];
    const float sa1 = m1 + sig1 * noise[i * 2 + 1];

    means[i * 2 + 0]  = m0;   means[i * 2 + 1]  = m1;
    sigmas[i * 2 + 0] = sig0; sigmas[i * 2 + 1] = sig1;
    sample[i * 2 + 0] = sa0;  sample[i * 2 + 1] = sa1;

    const float rng = (float)(HIc - HGc) - 1.f;   // 111
    pts[i * 2 + 0] = (int)rintf(rng / (1.f + expf(-sa0)));
    pts[i * 2 + 1] = (int)rintf(rng / (1.f + expf(-sa1)));
}

// ---------------------------------------------------------------------------
// Patch gather from the ORIGINAL f32 image (result must be exact f32 slices).
// ---------------------------------------------------------------------------
__global__ void extract_kernel(const float* __restrict__ image,
                               const int* __restrict__ pts,
                               float* __restrict__ out)
{
    const int bg = blockIdx.x;
    const int b  = bg / Gc;
    const int ph = pts[bg * 2 + 0];
    const int pw = pts[bg * 2 + 1];
    constexpr int PATCH = CIc * HGc * WGc;   // 768
    for (int i = threadIdx.x; i < PATCH; i += blockDim.x) {
        int t = i;
        const int ww = t % WGc; t /= WGc;
        const int hh = t % HGc;
        const int c  = t / HGc;
        out[bg * PATCH + i] =
            image[((b * CIc + c) * HIc + ph + hh) * WIc + pw + ww];
    }
}

// ---------------------------------------------------------------------------
// Host orchestration. Workspace (~310 MiB): bf16 image + bf16 weights +
// two 128 MiB bf16 activation ping-pong buffers + small linear buffers.
// ---------------------------------------------------------------------------
extern "C" void kernel_launch(void* const* d_in, const int* in_sizes, int n_in,
                              void* d_out, int out_size, void* d_ws, size_t ws_size,
                              hipStream_t stream)
{
    const float* image = (const float*)d_in[0];
    const float* noise = (const float*)d_in[1];
    const float* fw[9]  = { (const float*)d_in[2],  (const float*)d_in[4],
                            (const float*)d_in[6],  (const float*)d_in[8],
                            (const float*)d_in[10], (const float*)d_in[12],
                            (const float*)d_in[14], (const float*)d_in[16],
                            (const float*)d_in[18] };
    const float* bias[9] = { (const float*)d_in[3],  (const float*)d_in[5],
                             (const float*)d_in[7],  (const float*)d_in[9],
                             (const float*)d_in[11], (const float*)d_in[13],
                             (const float*)d_in[15], (const float*)d_in[17],
                             (const float*)d_in[19] };
    const int wsz[9] = { 32*27, 32*288, 64*288, 64*576, 128*576, 128*1152,
                         HIDc*HIDLIN, HIDLIN*HIDLIN, HIDLIN*Gc*3 };

    // ---- workspace suballocation (256B aligned) ----
    char* p = (char*)d_ws;
    auto take = [&](size_t bytes) -> char* {
        char* r = p; p += (bytes + 255) & ~(size_t)255; return r;
    };
    __bf16* bimg = (__bf16*)take((size_t)NB * CIc * HIc * WIc * 2);
    __bf16* bw[9];
    for (int i = 0; i < 9; ++i) bw[i] = (__bf16*)take((size_t)wsz[i] * 2);
    __bf16* buf0 = (__bf16*)take((size_t)NB * 32 * HIc * WIc * 2);  // 128 MiB
    __bf16* buf1 = (__bf16*)take((size_t)NB * 32 * HIc * WIc * 2);  // 128 MiB
    __bf16* lin1 = (__bf16*)take((size_t)NB * HIDLIN * 2);
    __bf16* lin2 = (__bf16*)take((size_t)NB * HIDLIN * 2);
    float*  lin3 = (float*) take((size_t)NB * Gc * 3 * 4);
    int*    pts  = (int*)   take((size_t)NB * Gc * 2 * 4);

    // ---- d_out layout: result | means | sigmas | sample ----
    float* result = (float*)d_out;
    float* means  = result + NB * Gc * CIc * HGc * WGc;
    float* sigmas = means  + NB * Gc * 2;
    float* sample = sigmas + NB * Gc * 2;

    const dim3 blk(256);
    auto cvt = [&](const float* s, __bf16* d, int n) {
        f32_to_bf16_kernel<<<(n + 255) / 256, blk, 0, stream>>>(s, d, n);
    };

    // ---- precision conversion (image + all weights) ----
    cvt(image, bimg, NB * CIc * HIc * WIc);
    for (int i = 0; i < 9; ++i) cvt(fw[i], bw[i], wsz[i]);

    // ---- conv stack ----
    conv3x3_relu_wmma<3, 32, 128, 128>
        <<<dim3(64, 1, NB), blk, 0, stream>>>(bimg, bw[0], bias[0], buf0);
    conv3x3_relu_wmma<32, 32, 128, 128>
        <<<dim3(64, 1, NB), blk, 0, stream>>>(buf0, bw[1], bias[1], buf1);
    {
        const int total = NB * 32 * 32 * 32;
        maxpool_kernel<32, 128, 128, 4>
            <<<(total + 255) / 256, blk, 0, stream>>>(buf1, buf0, total);
    }
    conv3x3_relu_wmma<32, 64, 32, 32>
        <<<dim3(4, 2, NB), blk, 0, stream>>>(buf0, bw[2], bias[2], buf1);
    conv3x3_relu_wmma<64, 64, 32, 32>
        <<<dim3(4, 2, NB), blk, 0, stream>>>(buf1, bw[3], bias[3], buf0);
    {
        const int total = NB * 64 * 16 * 16;
        maxpool_kernel<64, 32, 32, 2>
            <<<(total + 255) / 256, blk, 0, stream>>>(buf0, buf1, total);
    }
    conv3x3_relu_wmma<64, 128, 16, 16>
        <<<dim3(1, 4, NB), blk, 0, stream>>>(buf1, bw[4], bias[4], buf0);
    conv3x3_relu_wmma<128, 128, 16, 16>
        <<<dim3(1, 4, NB), blk, 0, stream>>>(buf0, bw[5], bias[5], buf1);

    // ---- linear stack ----
    gemm_bias_wmma<true, true>
        <<<dim3(4, 4), blk, 0, stream>>>(buf1, bw[6], bias[6], lin1,
                                         NB, HIDc, HIDLIN);
    gemm_bias_wmma<true, true>
        <<<dim3(4, 4), blk, 0, stream>>>(lin1, bw[7], bias[7], lin2,
                                         NB, HIDLIN, HIDLIN);
    gemm_bias_wmma<false, false>
        <<<dim3(1, 4), blk, 0, stream>>>(lin2, bw[8], bias[8], lin3,
                                         NB, HIDLIN, Gc * 3);

    // ---- head ----
    sample_kernel<<<(NB * Gc + 255) / 256, blk, 0, stream>>>(lin3, noise,
                                                             means, sigmas,
                                                             sample, pts);
    extract_kernel<<<NB * Gc, blk, 0, stream>>>(image, pts, result);
}
